// Hfsa_19988777796300
// MI455X (gfx1250) — matrix operand order repaired and checked by reading, not compiled, not run on hardware
//
#include <hip/hip_runtime.h>
#include <hip/hip_bf16.h>
#include <math.h>

// ---------------------------------------------------------------------------
// Types for CDNA5 WMMA (wave32): 16x16x32 f16 -> f32 accumulate
// ---------------------------------------------------------------------------
typedef _Float16 __attribute__((ext_vector_type(16))) v16h;
typedef _Float16 __attribute__((ext_vector_type(8)))  v8h;
typedef float    __attribute__((ext_vector_type(8)))  v8f;

#define QQ   65536
#define HH   64
#define WW   64
#define HW   4096

// ---------------------------------------------------------------------------
// Activation (compile-time selected)
// 0=none 1=gelu(tanh approx) 2=tanh 3=sigmoid 4=elu(x)+1
// ---------------------------------------------------------------------------
template<int ACT>
__device__ __forceinline__ float act_apply(float x) {
    if (ACT == 1) {
        float x3 = x * x * x;
        return 0.5f * x * (1.0f + tanhf(0.7978845608028654f * (x + 0.044715f * x3)));
    }
    if (ACT == 2) return tanhf(x);
    if (ACT == 3) return 1.0f / (1.0f + expf(-x));
    if (ACT == 4) return (x > 0.0f) ? (x + 1.0f) : expf(x);
    return x;
}

// ---------------------------------------------------------------------------
// WMMA GEMM:  C[M,Npad] = act( A[M,KP] * W[Npad,KP]^T + bias ) (+resid)
//   A f16 row-major (lda=KP, zero-padded), W f16 row-major [Npad,KP]
//   (zero-padded rows/cols).
// Tiling: one wave owns a 32-row M tile (two A fragments) x NT column tiles;
// Npad is covered by ngroups = Npad/(NT*16) groups. KP is a template constant
// so every fragment address is a single base register + 24-bit immediate.
// ---------------------------------------------------------------------------
template<int NT, int ACT, bool RES, int KP>
__global__ __launch_bounds__(256)
void k_gemm(const _Float16* __restrict__ A,
            const _Float16* __restrict__ Wm,
            const float* __restrict__ bias,
            const float* __restrict__ resid,
            _Float16* __restrict__ o16,
            float* __restrict__ o32,
            int M, int Nin, int Npad)
{
    int wave   = (int)((blockIdx.x * blockDim.x + threadIdx.x) >> 5);
    int wstep  = (int)((gridDim.x * blockDim.x) >> 5);
    int lane   = (int)(threadIdx.x & 31);
    int l15    = lane & 15;
    int mtiles  = M >> 5;                   // 32 rows per tile
    int ngroups = Npad / (NT * 16);
    int tiles   = mtiles * ngroups;

    for (int t = wave; t < tiles; t += wstep) {
        int mt = t / ngroups;
        int ng = t - mt * ngroups;
        int row0  = mt << 5;
        int ncol0 = ng * (NT * 16);

        v8f acc[2][NT];
#pragma unroll
        for (int h = 0; h < 2; ++h)
#pragma unroll
            for (int i = 0; i < NT; ++i)
#pragma unroll
                for (int j = 0; j < 8; ++j) acc[h][i][j] = 0.0f;

        // A fragment layout (CDNA5 ISA 7.12.2, 16-bit A 16x32):
        //   lane l (<16): row l, K {0..7,16..23}; lane l+16: row l, K {8..15,24..31}
        const int koffA = (lane < 16) ? 0 : 8;
        // B fragment (W^T, 32x16): lane l (<16): col n0+l, K {0..7,16..23};
        //   lane l+16: col n0+l-16, K {8..15,24..31}
        const int koffB = (lane < 16) ? 0 : 8;

        const _Float16* Ab = A  + (size_t)(row0 + l15)  * KP + koffA;
        const _Float16* Wb = Wm + (size_t)(ncol0 + l15) * KP + koffB;

        for (int k0 = 0; k0 < KP; k0 += 32) {
            v16h a0, a1;
            {
                v8h lo = *(const v8h*)(Ab);
                v8h hi = *(const v8h*)(Ab + 16);
#pragma unroll
                for (int i = 0; i < 8; ++i) { a0[i] = lo[i]; a0[i + 8] = hi[i]; }
            }
            {
                v8h lo = *(const v8h*)(Ab + 16 * KP);
                v8h hi = *(const v8h*)(Ab + 16 * KP + 16);
#pragma unroll
                for (int i = 0; i < 8; ++i) { a1[i] = lo[i]; a1[i + 8] = hi[i]; }
            }
#pragma unroll
            for (int nt = 0; nt < NT; ++nt) {
                v8h b0 = *(const v8h*)(Wb + (size_t)nt * 16 * KP);
                v8h b1 = *(const v8h*)(Wb + (size_t)nt * 16 * KP + 16);
                v16h b;
#pragma unroll
                for (int i = 0; i < 8; ++i) { b[i] = b0[i]; b[i + 8] = b1[i]; }
                acc[0][nt] = __builtin_amdgcn_wmma_f32_16x16x32_f16(
                    false, a0, false, b, (short)0, acc[0][nt], false, false);
                acc[1][nt] = __builtin_amdgcn_wmma_f32_16x16x32_f16(
                    false, a1, false, b, (short)0, acc[1][nt], false, false);
            }
            Ab += 32; Wb += 32;
        }

        // Epilogue: D layout: lanes 0-15 -> rows 0..7 in acc[j], lanes 16-31 -> 8..15
#pragma unroll
        for (int h = 0; h < 2; ++h) {
#pragma unroll
            for (int nt = 0; nt < NT; ++nt) {
                int n = ncol0 + (nt << 4) + l15;
#pragma unroll
                for (int j = 0; j < 8; ++j) {
                    int rrow = row0 + (h << 4) + ((lane < 16) ? j : (8 + j));
                    float v = acc[h][nt][j];
                    if (bias && n < Nin) v += bias[n];
                    v = act_apply<ACT>(v);
                    if (RES && n < Nin) v += resid[(size_t)rrow * Npad + n];
                    if (o16) o16[(size_t)rrow * Npad + n] = (_Float16)v;
                    if (o32) o32[(size_t)rrow * Npad + n] = v;
                }
            }
        }
    }
}

template<int NT, int ACT, bool RES, int KP>
static inline void gemm_launch(hipStream_t s,
                               const _Float16* A, const _Float16* Wm,
                               const float* bias, const float* resid,
                               _Float16* o16, float* o32,
                               int M, int Nin, int Npad)
{
    int tiles  = (M >> 5) * (Npad / (NT * 16));
    int blocks = (tiles + 7) / 8;          // 8 waves per 256-thread block
    if (blocks > 2048) blocks = 2048;
    k_gemm<NT, ACT, RES, KP><<<blocks, 256, 0, s>>>(A, Wm, bias, resid, o16, o32, M, Nin, Npad);
}

// ---------------------------------------------------------------------------
// Weight pack f32[N,K] -> f16[Npad,Kpad] zero-padded
// ---------------------------------------------------------------------------
__global__ void k_pack_w(const float* __restrict__ w, _Float16* __restrict__ o,
                         int N, int K, int Npad, int Kpad)
{
    int idx = blockIdx.x * blockDim.x + threadIdx.x;
    int tot = Npad * Kpad;
    if (idx >= tot) return;
    int r = idx / Kpad, c = idx - r * Kpad;
    o[idx] = (r < N && c < K) ? (_Float16)w[(size_t)r * K + c] : (_Float16)0.0f;
}

// ---------------------------------------------------------------------------
// Naive 3x3 SAME conv, NCHW, 64x64 spatial (negligible FLOPs)
// ---------------------------------------------------------------------------
__global__ void k_conv3(const float* __restrict__ in, const float* __restrict__ w,
                        const float* __restrict__ b, float* __restrict__ out,
                        int Cin, int Cout)
{
    int idx = blockIdx.x * blockDim.x + threadIdx.x;
    if (idx >= Cout * HW) return;
    int co = idx >> 12;
    int p  = idx & (HW - 1);
    int y  = p >> 6, x = p & 63;
    float acc = b[co];
    for (int ci = 0; ci < Cin; ++ci) {
        const float* ip = in + (size_t)ci * HW;
        const float* wp = w + ((size_t)co * Cin + ci) * 9;
        for (int dy = 0; dy < 3; ++dy) {
            int iy = y + dy - 1;
            if (iy < 0 || iy >= HH) continue;
            for (int dx = 0; dx < 3; ++dx) {
                int ix = x + dx - 1;
                if (ix < 0 || ix >= WW) continue;
                acc += wp[dy * 3 + dx] * ip[iy * WW + ix];
            }
        }
    }
    out[idx] = acc;
}

// ---------------------------------------------------------------------------
// Stage-1 per-query prep: constraint (bilinear border), bilinear taps for
// feat/ffeat (border=False), rel via nearest feat_coord, extras into tf16/tc16
// ---------------------------------------------------------------------------
__global__ void k_prep1(const float* __restrict__ coord, const float* __restrict__ cell,
                        const float* __restrict__ inp,
                        float* __restrict__ constraint,
                        int* __restrict__ idx4, float* __restrict__ w4,
                        float* __restrict__ relcell,
                        _Float16* __restrict__ tf16, _Float16* __restrict__ tc16)
{
    int q = blockIdx.x * blockDim.x + threadIdx.x;
    if (q >= QQ) return;
    float gy = coord[2 * q], gx = coord[2 * q + 1];

    float ix = ((gx + 1.0f) * WW - 1.0f) * 0.5f;
    float iy = ((gy + 1.0f) * HH - 1.0f) * 0.5f;
    float x0 = floorf(ix), y0 = floorf(iy);
    float wx1 = ix - x0, wy1 = iy - y0;
    float dys[2] = { y0, y0 + 1.0f }, wys[2] = { 1.0f - wy1, wy1 };
    float dxs[2] = { x0, x0 + 1.0f }, wxs[2] = { 1.0f - wx1, wx1 };

    float con0 = 0.f, con1 = 0.f, con2 = 0.f;
    int t = 0;
#pragma unroll
    for (int a = 0; a < 2; ++a)
#pragma unroll
        for (int bb = 0; bb < 2; ++bb) {
            float dy = dys[a], dx = dxs[bb];
            float wgt = wys[a] * wxs[bb];
            int yi = (int)fminf(fmaxf(dy, 0.f), 63.f);
            int xi = (int)fminf(fmaxf(dx, 0.f), 63.f);
            int id = yi * WW + xi;
            con0 += wgt * inp[0 * HW + id];
            con1 += wgt * inp[1 * HW + id];
            con2 += wgt * inp[2 * HW + id];
            float valid = (dx >= 0.f && dx < (float)WW && dy >= 0.f && dy < (float)HH) ? 1.f : 0.f;
            idx4[4 * q + t] = id;
            w4[4 * q + t]   = wgt * valid;
            ++t;
        }
    constraint[3 * q + 0] = con0;
    constraint[3 * q + 1] = con1;
    constraint[3 * q + 2] = con2;

    float rc0 = cell[2 * q] * (float)HH;
    float rc1 = cell[2 * q + 1] * (float)WW;
    relcell[2 * q] = rc0; relcell[2 * q + 1] = rc1;

    float rel0 = 0.f, rel1 = 0.f;
#pragma unroll
    for (int tt = 0; tt < 2; ++tt) {
        float d  = (tt == 0) ? (-1.0f / HH) : (1.0f / HH);
        float cy = fminf(fmaxf(gy + d + 1e-6f, -1.f + 1e-6f), 1.f - 1e-6f);
        float cx = fminf(fmaxf(gx + d + 1e-6f, -1.f + 1e-6f), 1.f - 1e-6f);
        float fxn = floorf(((cx + 1.f) * WW - 1.f) * 0.5f + 0.5f);
        float fyn = floorf(((cy + 1.f) * HH - 1.f) * 0.5f + 0.5f);
        float valid = (fxn >= 0.f && fxn < (float)WW && fyn >= 0.f && fyn < (float)HH) ? 1.f : 0.f;
        float yc = fminf(fmaxf(fyn, 0.f), 63.f);
        float xc = fminf(fmaxf(fxn, 0.f), 63.f);
        float qc0 = (-1.f + 1.f / HH + (2.f / HH) * yc) * valid;
        float qc1 = (-1.f + 1.f / WW + (2.f / WW) * xc) * valid;
        rel0 += (gy - qc0) * (float)HH;
        rel1 += (gx - qc1) * (float)WW;
    }
    rel0 *= 0.5f; rel1 *= 0.5f;

    _Float16* tfr = tf16 + (size_t)q * 288;
    tfr[256] = (_Float16)rc0;  tfr[257] = (_Float16)rc1;
    tfr[258] = (_Float16)rel0; tfr[259] = (_Float16)rel1;
    for (int c = 260; c < 288; ++c) tfr[c] = (_Float16)0.f;

    _Float16* tcr = tc16 + (size_t)q * 96;
    tcr[64] = (_Float16)rc0;  tcr[65] = (_Float16)rc1;
    tcr[66] = (_Float16)rel0; tcr[67] = (_Float16)rel1;
    for (int c = 68; c < 96; ++c) tcr[c] = (_Float16)0.f;
}

// 4-tap weighted gather into padded f16 activation rows
__global__ void k_gather4(const float* __restrict__ src, int C,
                          const int* __restrict__ idx4, const float* __restrict__ w4,
                          _Float16* __restrict__ dst, int ldd)
{
    long long tid = (long long)blockIdx.x * blockDim.x + threadIdx.x;
    if (tid >= (long long)QQ * C) return;
    int q = (int)(tid / C), c = (int)(tid - (long long)q * C);
    const float* fc = src + (size_t)c * HW;
    float acc = 0.f;
#pragma unroll
    for (int j = 0; j < 4; ++j) acc += w4[4 * q + j] * fc[idx4[4 * q + j]];
    dst[(size_t)q * ldd + c] = (_Float16)acc;
}

// ---------------------------------------------------------------------------
// Stage-2: flow update + 4-tap area weights + fusion extras (relcell + code0)
// ---------------------------------------------------------------------------
__global__ void k_prep2(const float* __restrict__ coord,
                        const float* __restrict__ flow, const float* __restrict__ cg,
                        const float* __restrict__ relcell,
                        int* __restrict__ idx2, float* __restrict__ avw,
                        float* __restrict__ ratio, _Float16* __restrict__ tfus)
{
    int q = blockIdx.x * blockDim.x + threadIdx.x;
    if (q >= QQ) return;
    float gy = coord[2 * q]     + (flow[q * 16 + 0] / (float)HH) * cg[q * 16 + 0];
    float gx = coord[2 * q + 1] + (flow[q * 16 + 1] / (float)WW) * cg[q * 16 + 1];

    const float ry = 1.0f / HH, rx = 1.0f / WW;
    const float dy4[4] = { -ry,  ry,  ry, -ry };
    const float dx4[4] = { -rx, -rx,  rx,  rx };

    float rel0[4], rel1[4], areas[4], valid[4];
#pragma unroll
    for (int j = 0; j < 4; ++j) {
        float cy = fminf(fmaxf(gy + dy4[j] + 1e-6f, -1.f + 1e-6f), 1.f - 1e-6f);
        float cx = fminf(fmaxf(gx + dx4[j] + 1e-6f, -1.f + 1e-6f), 1.f - 1e-6f);
        float fxn = floorf(((cx + 1.f) * WW - 1.f) * 0.5f + 0.5f);
        float fyn = floorf(((cy + 1.f) * HH - 1.f) * 0.5f + 0.5f);
        float v = (fxn >= 0.f && fxn < (float)WW && fyn >= 0.f && fyn < (float)HH) ? 1.f : 0.f;
        float yc = fminf(fmaxf(fyn, 0.f), 63.f);
        float xc = fminf(fmaxf(fxn, 0.f), 63.f);
        idx2[4 * q + j] = (int)yc * WW + (int)xc;
        float qc0 = (-1.f + 1.f / HH + (2.f / HH) * yc) * v;
        float qc1 = (-1.f + 1.f / WW + (2.f / WW) * xc) * v;
        rel0[j] = (gy - qc0) * (float)HH;
        rel1[j] = (gx - qc1) * (float)WW;
        areas[j] = fabsf(rel0[j] * rel1[j]) + 1e-9f;
        valid[j] = v;
    }
    float asum = areas[0] + areas[1] + areas[2] + areas[3] + 1e-9f;
#pragma unroll
    for (int j = 0; j < 4; ++j)
        avw[4 * q + j] = (areas[3 - j] / asum) * valid[j];   // reversed-area weights

    float r0 = (rel0[0] + rel0[1] + rel0[2] + rel0[3]) * 0.25f;
    float r1 = (rel1[0] + rel1[1] + rel1[2] + rel1[3]) * 0.25f;
    ratio[2 * q] = r0; ratio[2 * q + 1] = r1;

    const float PI = 3.14159265358979323846f;
    _Float16* tr = tfus + (size_t)q * 1056;
    tr[1024] = (_Float16)relcell[2 * q];
    tr[1025] = (_Float16)relcell[2 * q + 1];
    tr[1026] = (_Float16)sinf(PI * r0);
    tr[1027] = (_Float16)sinf(PI * r1);
    tr[1028] = (_Float16)cosf(PI * r0);
    tr[1029] = (_Float16)cosf(PI * r1);
    for (int c = 1030; c < 1056; ++c) tr[c] = (_Float16)0.f;
}

// q_ffeat * areas gather: tfus[q, c*4+j] = ffeat[c, idx_j] * avw_j
__global__ void k_gather_fus(const float* __restrict__ ffeat,
                             const int* __restrict__ idx2, const float* __restrict__ avw,
                             _Float16* __restrict__ tfus)
{
    long long tid = (long long)blockIdx.x * blockDim.x + threadIdx.x;
    if (tid >= (long long)QQ * 256) return;
    int q = (int)(tid >> 8), c = (int)(tid & 255);
    const float* fc = ffeat + (size_t)c * HW;
    _Float16* tr = tfus + (size_t)q * 1056 + (size_t)c * 4;
#pragma unroll
    for (int j = 0; j < 4; ++j)
        tr[j] = (_Float16)(fc[idx2[4 * q + j]] * avw[4 * q + j]);
}

// ---------------------------------------------------------------------------
// Linear attention helpers
// ---------------------------------------------------------------------------
__global__ void k_zero_f32(float* p, int n)
{
    int i = blockIdx.x * blockDim.x + threadIdx.x;
    if (i < n) p[i] = 0.f;
}

// kv[h,d,e] = sum_q k[q,hd]*v[q,he]; ksum[h,d] = sum_q k[q,hd]
__global__ void k_kv_reduce(const float* __restrict__ ka, const float* __restrict__ va,
                            float* __restrict__ kv, float* __restrict__ ksum)
{
    __shared__ float sk[256], sv[256];
    int t = threadIdx.x;
    int h = t >> 4, d = t & 15;
    float acc[16];
#pragma unroll
    for (int e = 0; e < 16; ++e) acc[e] = 0.f;
    float accs = 0.f;
    for (int q = blockIdx.x; q < QQ; q += gridDim.x) {
        sk[t] = ka[(size_t)q * 256 + t];
        sv[t] = va[(size_t)q * 256 + t];
        __syncthreads();
        float kval = sk[h * 16 + d];
        accs += kval;
#pragma unroll
        for (int e = 0; e < 16; ++e) acc[e] += kval * sv[h * 16 + e];
        __syncthreads();
    }
#pragma unroll
    for (int e = 0; e < 16; ++e) atomicAdd(&kv[(h * 16 + d) * 16 + e], acc[e]);
    atomicAdd(&ksum[h * 16 + d], accs);
}

__global__ void k_atten_combine(const float* __restrict__ qa,
                                const float* __restrict__ kv, const float* __restrict__ ksum,
                                _Float16* __restrict__ out16)
{
    long long tid = (long long)blockIdx.x * blockDim.x + threadIdx.x;
    if (tid >= (long long)QQ * 16) return;
    int q = (int)(tid >> 4), h = (int)(tid & 15);
    const float* qr = qa + (size_t)q * 256 + h * 16;
    float den = 1e-6f;
#pragma unroll
    for (int d = 0; d < 16; ++d) den += qr[d] * ksum[h * 16 + d];
    float z = 1.0f / den;
#pragma unroll
    for (int e = 0; e < 16; ++e) {
        float o = 0.f;
#pragma unroll
        for (int d = 0; d < 16; ++d) o += qr[d] * kv[(h * 16 + d) * 16 + e];
        out16[(size_t)q * 256 + h * 16 + e] = (_Float16)(o * z);
    }
}

// imnet input builder: [grid16 | fourier(ratio, w) | zeros] -> t16[Q,288]
__global__ void k_build_t(const _Float16* __restrict__ g16, const float* __restrict__ ratio,
                          float wfreq, _Float16* __restrict__ t16)
{
    long long tid = (long long)blockIdx.x * blockDim.x + threadIdx.x;
    if (tid >= (long long)QQ * 288) return;
    int q = (int)(tid / 288), c = (int)(tid - (long long)q * 288);
    _Float16 v;
    if (c < 256)      v = g16[(size_t)q * 256 + c];
    else if (c < 260) {
        int j = c - 256;
        float r = ratio[2 * q + (j & 1)];
        v = (_Float16)((j < 2) ? sinf(wfreq * r) : cosf(wfreq * r));
    } else v = (_Float16)0.f;
    t16[tid] = v;
}

__global__ void k_final(const float* __restrict__ outf, const float* __restrict__ constraint,
                        float* __restrict__ out)
{
    long long tid = (long long)blockIdx.x * blockDim.x + threadIdx.x;
    if (tid >= (long long)QQ * 3) return;
    int q = (int)(tid / 3), c = (int)(tid - (long long)q * 3);
    out[tid] = outf[(size_t)q * 16 + c] + constraint[tid];
}

// ---------------------------------------------------------------------------
// Host driver
// ---------------------------------------------------------------------------
extern "C" void kernel_launch(void* const* d_in, const int* in_sizes, int n_in,
                              void* d_out, int out_size, void* d_ws, size_t ws_size,
                              hipStream_t stream)
{
    (void)in_sizes; (void)n_in; (void)out_size; (void)ws_size;
    char* ws = (char*)d_ws;
    size_t off = 0;
    auto alloc = [&](size_t bytes) -> void* {
        size_t o = (off + 255) & ~(size_t)255;
        off = o + bytes;
        return (void*)(ws + o);
    };

    const float* inp   = (const float*)d_in[0];
    const float* coord = (const float*)d_in[1];
    const float* cell  = (const float*)d_in[2];

    // JAX pytree (dict keys sorted) flat leaf order:
    // atten1{k,o,q,v}{b,w}=3..10  atten2=11..18  enc{b,w}=19,20  expan{b,w}=21,22
    // flow_c[0..2]{b,w}=23..28    flow_f=29..34
    // fusion{f[0..2],s}{b,w}=35..42
    // imnet[i]{f0b,f0w,f1b,f1w,f2b,f2w,sb,sw}=43+8i
    auto B = [&](int i) { return (const float*)d_in[i]; };

    auto pack = [&](int i, int N, int K, int Npad, int Kpad) -> const _Float16* {
        _Float16* p = (_Float16*)alloc((size_t)Npad * Kpad * sizeof(_Float16));
        int tot = Npad * Kpad;
        k_pack_w<<<(tot + 255) / 256, 256, 0, stream>>>((const float*)d_in[i], p, N, K, Npad, Kpad);
        return p;
    };

    // ---- pack all weights to padded f16 ----
    const _Float16* Wk[2] = { pack(4, 256, 256, 256, 256),  pack(12, 256, 256, 256, 256) };
    const _Float16* Wo[2] = { pack(6, 256, 256, 256, 256),  pack(14, 256, 256, 256, 256) };
    const _Float16* Wq[2] = { pack(8, 256, 256, 256, 256),  pack(16, 256, 256, 256, 256) };
    const _Float16* Wv[2] = { pack(10, 256, 256, 256, 256), pack(18, 256, 256, 256, 256) };
    const float* bk[2] = { B(3),  B(11) };
    const float* bo[2] = { B(5),  B(13) };
    const float* bq[2] = { B(7),  B(15) };
    const float* bv[2] = { B(9),  B(17) };

    const _Float16* Wfc0 = pack(24, 64, 68, 64, 96);
    const _Float16* Wfc1 = pack(26, 64, 64, 64, 64);
    const _Float16* Wfc2 = pack(28, 2, 64, 16, 64);
    const _Float16* Wff0 = pack(30, 256, 260, 256, 288);
    const _Float16* Wff1 = pack(32, 256, 256, 256, 256);
    const _Float16* Wff2 = pack(34, 2, 256, 16, 256);

    const _Float16* Wfu0 = pack(36, 256, 1030, 256, 1056);
    const _Float16* Wfu1 = pack(38, 256, 256, 256, 256);
    const _Float16* Wfu2 = pack(40, 256, 256, 256, 256);
    const _Float16* Wfus = pack(42, 256, 1030, 256, 1056);

    const _Float16* Wi0[4]; const _Float16* Wi1[4]; const _Float16* Wi2[4]; const _Float16* Wis[4];
    for (int i = 0; i < 4; ++i) {
        int base = 43 + 8 * i;
        int nout = (i == 3) ? 3 : 256;
        int npad = (i == 3) ? 16 : 256;
        Wi0[i] = pack(base + 1, 256, 260, 256, 288);
        Wi1[i] = pack(base + 3, 256, 256, 256, 256);
        Wi2[i] = pack(base + 5, nout, 256, npad, 256);
        Wis[i] = pack(base + 7, nout, 260, npad, 288);
    }

    // ---- activation / scratch buffers ----
    float*    feat   = (float*)alloc((size_t)64 * HW * 4);
    float*    ffeat  = (float*)alloc((size_t)256 * HW * 4);
    float*    constr = (float*)alloc((size_t)QQ * 3 * 4);
    int*      idx4   = (int*)alloc((size_t)QQ * 4 * 4);
    float*    w4     = (float*)alloc((size_t)QQ * 4 * 4);
    float*    relcl  = (float*)alloc((size_t)QQ * 2 * 4);
    _Float16* tf16   = (_Float16*)alloc((size_t)QQ * 288 * 2);   // also reused as imnet t16
    _Float16* tc16   = (_Float16*)alloc((size_t)QQ * 96 * 2);
    _Float16* h16a   = (_Float16*)alloc((size_t)QQ * 256 * 2);   // also attention out16
    _Float16* h16b   = (_Float16*)alloc((size_t)QQ * 256 * 2);
    _Float16* hc16a  = (_Float16*)alloc((size_t)QQ * 64 * 2);
    _Float16* hc16b  = (_Float16*)alloc((size_t)QQ * 64 * 2);
    float*    flowv  = (float*)alloc((size_t)QQ * 16 * 4);
    float*    cgate  = (float*)alloc((size_t)QQ * 16 * 4);
    int*      idx2   = (int*)alloc((size_t)QQ * 4 * 4);
    float*    avw    = (float*)alloc((size_t)QQ * 4 * 4);
    float*    ratio  = (float*)alloc((size_t)QQ * 2 * 4);
    _Float16* tfus16 = (_Float16*)alloc((size_t)QQ * 1056 * 2);  // reused as qa/ka (f32)
    float*    sres   = (float*)alloc((size_t)QQ * 256 * 4);      // reused as va (f32)
    float*    grid32 = (float*)alloc((size_t)QQ * 256 * 4);
    _Float16* grid16 = (_Float16*)alloc((size_t)QQ * 256 * 2);
    float*    kvbuf  = (float*)alloc((size_t)(4096 + 256) * 4);
    float*    sres16 = (float*)alloc((size_t)QQ * 16 * 4);
    float*    outf   = (float*)alloc((size_t)QQ * 16 * 4);

    const int M = QQ;

    // ---- convolutions (tiny) ----
    k_conv3<<<(64 * HW + 255) / 256, 256, 0, stream>>>(inp, B(20), B(19), feat, 3, 64);
    k_conv3<<<(256 * HW + 255) / 256, 256, 0, stream>>>(feat, B(22), B(21), ffeat, 64, 256);

    // ---- stage 1: per-query prep + gathers ----
    k_prep1<<<(QQ + 255) / 256, 256, 0, stream>>>(coord, cell, inp, constr, idx4, w4, relcl, tf16, tc16);
    k_gather4<<<(int)(((long long)QQ * 64 + 255) / 256), 256, 0, stream>>>(feat, 64, idx4, w4, tc16, 96);
    k_gather4<<<(int)(((long long)QQ * 256 + 255) / 256), 256, 0, stream>>>(ffeat, 256, idx4, w4, tf16, 288);

    // ---- flow_f MLP (260->256->256->2 tanh) ----
    gemm_launch<8, 1, false, 288>(stream, tf16, Wff0, B(29), nullptr, h16a, nullptr, M, 256, 256);
    gemm_launch<8, 1, false, 256>(stream, h16a, Wff1, B(31), nullptr, h16b, nullptr, M, 256, 256);
    gemm_launch<1, 2, false, 256>(stream, h16b, Wff2, B(33), nullptr, nullptr, flowv, M, 2, 16);
    // ---- flow_c MLP (68->64->64->2 sigmoid) ----
    gemm_launch<4, 1, false, 96>(stream, tc16, Wfc0, B(23), nullptr, hc16a, nullptr, M, 64, 64);
    gemm_launch<4, 1, false, 64>(stream, hc16a, Wfc1, B(25), nullptr, hc16b, nullptr, M, 64, 64);
    gemm_launch<1, 3, false, 64>(stream, hc16b, Wfc2, B(27), nullptr, nullptr, cgate, M, 2, 16);

    // ---- stage 2: coord update + area gather ----
    k_prep2<<<(QQ + 255) / 256, 256, 0, stream>>>(coord, flowv, cgate, relcl, idx2, avw, ratio, tfus16);
    k_gather_fus<<<(int)(((long long)QQ * 256 + 255) / 256), 256, 0, stream>>>(ffeat, idx2, avw, tfus16);

    // ---- fusion mlp_sc (1030 -> 256, skip) ----
    gemm_launch<8, 0, false, 1056>(stream, tfus16, Wfus, B(41), nullptr, nullptr, sres, M, 256, 256);
    gemm_launch<8, 1, false, 1056>(stream, tfus16, Wfu0, B(35), nullptr, h16a, nullptr, M, 256, 256);
    gemm_launch<8, 1, false, 256>(stream, h16a, Wfu1, B(37), nullptr, h16b, nullptr, M, 256, 256);
    gemm_launch<8, 0, true, 256>(stream, h16b, Wfu2, B(39), sres, grid16, grid32, M, 256, 256);

    // ---- linear attention x2 (elu+1 fused in q/k epilogue) ----
    float* qa = (float*)tfus16;                       // reuse (64 MB)
    float* ka = (float*)((char*)tfus16 + (size_t)QQ * 256 * 4);
    float* va = sres;                                 // reuse (64 MB)
    float* kv = kvbuf;
    float* ksum = kvbuf + 4096;
    for (int a = 0; a < 2; ++a) {
        gemm_launch<8, 4, false, 256>(stream, grid16, Wq[a], bq[a], nullptr, nullptr, qa, M, 256, 256);
        gemm_launch<8, 4, false, 256>(stream, grid16, Wk[a], bk[a], nullptr, nullptr, ka, M, 256, 256);
        gemm_launch<8, 0, false, 256>(stream, grid16, Wv[a], bv[a], nullptr, nullptr, va, M, 256, 256);
        k_zero_f32<<<(4352 + 255) / 256, 256, 0, stream>>>(kvbuf, 4352);
        k_kv_reduce<<<256, 256, 0, stream>>>(ka, va, kv, ksum);
        k_atten_combine<<<(int)(((long long)QQ * 16 + 255) / 256), 256, 0, stream>>>(qa, kv, ksum, h16a);
        gemm_launch<8, 0, true, 256>(stream, h16a, Wo[a], bo[a], grid32, grid16, grid32, M, 256, 256);
    }

    // ---- imnet layers ----
    _Float16* t16 = tf16;                             // reuse (same [Q,288] shape)
    for (int i = 0; i < 4; ++i) {
        float wfreq = 3.14159265358979323846f * (float)(1 << (i + 1));
        k_build_t<<<(int)(((long long)QQ * 288 + 255) / 256), 256, 0, stream>>>(grid16, ratio, wfreq, t16);
        int base = 43 + 8 * i;
        if (i < 3) {
            gemm_launch<8, 0, false, 288>(stream, t16, Wis[i], B(base + 6), nullptr, nullptr, sres, M, 256, 256);
            gemm_launch<8, 1, false, 288>(stream, t16, Wi0[i], B(base + 0), nullptr, h16a, nullptr, M, 256, 256);
            gemm_launch<8, 1, false, 256>(stream, h16a, Wi1[i], B(base + 2), nullptr, h16b, nullptr, M, 256, 256);
            gemm_launch<8, 0, true, 256>(stream, h16b, Wi2[i], B(base + 4), sres, grid16, grid32, M, 256, 256);
        } else {
            gemm_launch<1, 0, false, 288>(stream, t16, Wis[i], B(base + 6), nullptr, nullptr, sres16, M, 3, 16);
            gemm_launch<8, 1, false, 288>(stream, t16, Wi0[i], B(base + 0), nullptr, h16a, nullptr, M, 256, 256);
            gemm_launch<8, 1, false, 256>(stream, h16a, Wi1[i], B(base + 2), nullptr, h16b, nullptr, M, 256, 256);
            gemm_launch<1, 0, true, 256>(stream, h16b, Wi2[i], B(base + 4), sres16, nullptr, outf, M, 3, 16);
        }
    }

    // ---- final: add bilinear constraint ----
    k_final<<<(int)(((long long)QQ * 3 + 255) / 256), 256, 0, stream>>>(outf, constr, (float*)d_out);
}
